// NodeAttentionHead_35871566856204
// MI455X (gfx1250) — matrix operand order repaired and checked by reading, not compiled, not run on hardware
//
#include <hip/hip_runtime.h>
#include <hip/hip_bf16.h>

typedef __attribute__((ext_vector_type(16))) _Float16 v16h;
typedef __attribute__((ext_vector_type(8)))  _Float16 v8h;
typedef __attribute__((ext_vector_type(8)))  float    v8f;

#define N_NODES 50000
#define N_EDGES 400000
#define TWO_E   800000

// ---------------- workspace layout (bytes) ----------------
#define O_HV    0UL          // h_v              N*128*4 = 25,600,000
#define O_EV    25600000UL   // e_v rows [0,N)   N*128*4 = 25,600,000
#define O_T     51200000UL   // t = e_v . a_edge[128:]   E*4
#define O_S1    52800000UL   // h_v . a_node[:128]       N*4
#define O_S2    53000000UL   // h_v . a_node[128:]       N*4
#define O_S3    53200000UL   // h_v . a_edge[:128]       N*4
#define O_NATT  53400000UL   // 2E*4
#define O_EATT  56600000UL   // 2E*4
#define O_NDEN  59800000UL   // 2E*4
#define O_EDEN  63000000UL   // 2E*4
#define O_NSUM  66200000UL   // N*4   <-- zero region starts here
#define O_ESUM  66400000UL   // N*4
#define O_CNT   66600000UL   // N*4 (int)
#define O_ACC   66800000UL   // 4 doubles (8-aligned)
#define ZERO_BYTES 600032UL  // [O_NSUM, O_ACC+32)
#define O_PREF  66800032UL   // (N)*4 int

// =====================================================================
// GEMM 1: hv[N,128] = node_fts[N,256] @ W_node[256,128]  (WMMA f16->f32)
// One wave = 16 rows x 128 cols; W staged transposed in LDS as f16.
// =====================================================================
__global__ void __launch_bounds__(256)
k_gemm_node(const float* __restrict__ A, const float* __restrict__ W,
            float* __restrict__ hv) {
  __shared__ _Float16 Wt[128 * 256];   // Wt[col*256 + k], 64KB
  int tid = threadIdx.x;
  for (int it = 0; it < 128; ++it) {
    int idx = it * 256 + tid;          // 32768 elements
    int k = idx >> 7, col = idx & 127; // W row-major [k,col]
    Wt[col * 256 + k] = (_Float16)W[idx];
  }
  __syncthreads();
  int wave = tid >> 5, lane = tid & 31;
  int rt = blockIdx.x * 8 + wave;
  if (rt >= N_NODES / 16) return;
  int half = lane >> 4;                // 0: lanes 0-15, 1: lanes 16-31
  int lrow = lane & 15;
  const float* Arow = A + (rt * 16 + lrow) * 256;
  v8f zero = {0.f,0.f,0.f,0.f,0.f,0.f,0.f,0.f};
  v8f acc[8];
#pragma unroll
  for (int n = 0; n < 8; ++n) acc[n] = zero;
#pragma unroll
  for (int kb = 0; kb < 8; ++kb) {
    // A fragment: lane half selects K = base+{0..7} and base+16+{0..7}
    int ka = kb * 32 + half * 8;
    float4 a0 = *(const float4*)(Arow + ka);
    float4 a1 = *(const float4*)(Arow + ka + 4);
    float4 a2 = *(const float4*)(Arow + ka + 16);
    float4 a3 = *(const float4*)(Arow + ka + 20);
    v16h af;
    af[0]=(_Float16)a0.x; af[1]=(_Float16)a0.y; af[2]=(_Float16)a0.z; af[3]=(_Float16)a0.w;
    af[4]=(_Float16)a1.x; af[5]=(_Float16)a1.y; af[6]=(_Float16)a1.z; af[7]=(_Float16)a1.w;
    af[8]=(_Float16)a2.x; af[9]=(_Float16)a2.y; af[10]=(_Float16)a2.z; af[11]=(_Float16)a2.w;
    af[12]=(_Float16)a3.x; af[13]=(_Float16)a3.y; af[14]=(_Float16)a3.z; af[15]=(_Float16)a3.w;
    // B fragment: lane half selects K = base+{0..7} and base+16+{0..7}
    int kc = kb * 32 + half * 8;
#pragma unroll
    for (int n = 0; n < 8; ++n) {
      int col = n * 16 + lrow;
      v8h b0 = *(const v8h*)&Wt[col * 256 + kc];
      v8h b1 = *(const v8h*)&Wt[col * 256 + kc + 16];
      v16h bb;
#pragma unroll
      for (int j = 0; j < 8; ++j) { bb[j] = b0[j]; bb[8 + j] = b1[j]; }
      acc[n] = __builtin_amdgcn_wmma_f32_16x16x32_f16(
          false, af, false, bb, (short)0, acc[n], false, false);
    }
  }
  // C/D layout: VGPR v -> M = v + half*8, N = lrow (per col-tile)
  int mbase = rt * 16 + half * 8;
#pragma unroll
  for (int n = 0; n < 8; ++n) {
    int col = n * 16 + lrow;
#pragma unroll
    for (int v = 0; v < 8; ++v)
      hv[(mbase + v) * 128 + col] = acc[n][v];
  }
}

// =====================================================================
// GEMM 2: e_v[E,128] = edge_fts @ W_edge; store only rows < N (the only
// rows ever gathered, since dst indices < N); epilogue computes
// t[row] = e_v[row] . a_edge[128:256] via half-wave shuffle reduction.
// =====================================================================
__global__ void __launch_bounds__(256)
k_gemm_edge(const float* __restrict__ A, const float* __restrict__ W,
            const float* __restrict__ a_edge, float* __restrict__ evN,
            float* __restrict__ t) {
  __shared__ _Float16 Wt[128 * 128];   // Wt[col*128 + k], 32KB
  int tid = threadIdx.x;
  for (int it = 0; it < 64; ++it) {
    int idx = it * 256 + tid;          // 16384 elements
    int k = idx >> 7, col = idx & 127;
    Wt[col * 128 + k] = (_Float16)W[idx];
  }
  __syncthreads();
  int wave = tid >> 5, lane = tid & 31;
  int rt = blockIdx.x * 8 + wave;      // 25000 tiles exactly
  int half = lane >> 4;
  int lrow = lane & 15;
  const float* Arow = A + (rt * 16 + lrow) * 128;
  float aeL[8];
#pragma unroll
  for (int n = 0; n < 8; ++n) aeL[n] = a_edge[128 + n * 16 + lrow];
  v8f zero = {0.f,0.f,0.f,0.f,0.f,0.f,0.f,0.f};
  v8f acc[8];
#pragma unroll
  for (int n = 0; n < 8; ++n) acc[n] = zero;
#pragma unroll
  for (int kb = 0; kb < 4; ++kb) {
    int ka = kb * 32 + half * 8;
    float4 a0 = *(const float4*)(Arow + ka);
    float4 a1 = *(const float4*)(Arow + ka + 4);
    float4 a2 = *(const float4*)(Arow + ka + 16);
    float4 a3 = *(const float4*)(Arow + ka + 20);
    v16h af;
    af[0]=(_Float16)a0.x; af[1]=(_Float16)a0.y; af[2]=(_Float16)a0.z; af[3]=(_Float16)a0.w;
    af[4]=(_Float16)a1.x; af[5]=(_Float16)a1.y; af[6]=(_Float16)a1.z; af[7]=(_Float16)a1.w;
    af[8]=(_Float16)a2.x; af[9]=(_Float16)a2.y; af[10]=(_Float16)a2.z; af[11]=(_Float16)a2.w;
    af[12]=(_Float16)a3.x; af[13]=(_Float16)a3.y; af[14]=(_Float16)a3.z; af[15]=(_Float16)a3.w;
    int kc = kb * 32 + half * 8;
#pragma unroll
    for (int n = 0; n < 8; ++n) {
      int col = n * 16 + lrow;
      v8h b0 = *(const v8h*)&Wt[col * 128 + kc];
      v8h b1 = *(const v8h*)&Wt[col * 128 + kc + 16];
      v16h bb;
#pragma unroll
      for (int j = 0; j < 8; ++j) { bb[j] = b0[j]; bb[8 + j] = b1[j]; }
      acc[n] = __builtin_amdgcn_wmma_f32_16x16x32_f16(
          false, af, false, bb, (short)0, acc[n], false, false);
    }
  }
  int mbase = rt * 16 + half * 8;
  // t epilogue: row M = mbase+v; its cols live in lanes of the same half
#pragma unroll
  for (int v = 0; v < 8; ++v) {
    float p = 0.f;
#pragma unroll
    for (int n = 0; n < 8; ++n) p += acc[n][v] * aeL[n];
    p += __shfl_xor(p, 1); p += __shfl_xor(p, 2);
    p += __shfl_xor(p, 4); p += __shfl_xor(p, 8);
    if (lrow == 0) t[mbase + v] = p;
  }
  // store only rows used by gathers (row index is a node id < N)
#pragma unroll
  for (int n = 0; n < 8; ++n) {
    int col = n * 16 + lrow;
#pragma unroll
    for (int v = 0; v < 8; ++v) {
      int r = mbase + v;
      if (r < N_NODES) evN[r * 128 + col] = acc[n][v];
    }
  }
}

// ---------------- per-node attention score dots ----------------
__global__ void __launch_bounds__(256)
k_scores(const float* __restrict__ hv, const float* __restrict__ a_node,
         const float* __restrict__ a_edge, float* __restrict__ s1,
         float* __restrict__ s2, float* __restrict__ s3) {
  int wave = threadIdx.x >> 5, lane = threadIdx.x & 31;
  int row = blockIdx.x * 8 + wave;
  if (row >= N_NODES) return;
  float4 h   = *(const float4*)(hv + row * 128 + lane * 4);
  float4 an1 = *(const float4*)(a_node + lane * 4);
  float4 an2 = *(const float4*)(a_node + 128 + lane * 4);
  float4 ae1 = *(const float4*)(a_edge + lane * 4);
  float p1 = h.x*an1.x + h.y*an1.y + h.z*an1.z + h.w*an1.w;
  float p2 = h.x*an2.x + h.y*an2.y + h.z*an2.z + h.w*an2.w;
  float p3 = h.x*ae1.x + h.y*ae1.y + h.z*ae1.z + h.w*ae1.w;
#pragma unroll
  for (int m = 16; m > 0; m >>= 1) {
    p1 += __shfl_xor(p1, m); p2 += __shfl_xor(p2, m); p3 += __shfl_xor(p3, m);
  }
  if (lane == 0) { s1[row] = p1; s2[row] = p2; s3[row] = p3; }
}

// ---------------- per-edge attention + segment sums + bincount ----------------
__global__ void __launch_bounds__(256)
k_att(const int* __restrict__ eb, const float* __restrict__ s1,
      const float* __restrict__ s2, const float* __restrict__ s3,
      const float* __restrict__ t, float* __restrict__ natt,
      float* __restrict__ eatt, float* __restrict__ nsum,
      float* __restrict__ esum, int* __restrict__ cnt) {
  int i = blockIdx.x * 256 + threadIdx.x;   // exact 800000
  int s, d, ii;
  if (i < N_EDGES) { ii = i;           s = eb[2*ii];     d = eb[2*ii + 1]; }
  else             { ii = i - N_EDGES; s = eb[2*ii + 1]; d = eb[2*ii];     }
  float xn = s1[s] + s2[d];
  xn = (xn >= 0.f) ? xn : 0.2f * xn;
  xn = fminf(2.f, fmaxf(-2.f, xn));
  float na = expf(xn);
  float xe = s3[s] + t[ii];
  xe = (xe >= 0.f) ? xe : 0.2f * xe;
  xe = fminf(2.f, fmaxf(-2.f, xe));
  float ea = expf(xe);
  natt[i] = na; eatt[i] = ea;
  atomicAdd(&nsum[s], na);
  atomicAdd(&esum[s], ea);
  atomicAdd(&cnt[s], 1);
}

// ---------------- exclusive prefix scan of counts (single block) ----------------
__global__ void __launch_bounds__(1024)
k_scan(const int* __restrict__ cnt, int* __restrict__ prefix) {
  __shared__ int buf[1024];
  __shared__ int carry;
  if (threadIdx.x == 0) carry = 0;
  __syncthreads();
  for (int base = 0; base < N_NODES; base += 1024) {
    int idx = base + threadIdx.x;
    int v = (idx < N_NODES) ? cnt[idx] : 0;
    buf[threadIdx.x] = v;
    __syncthreads();
    for (int off = 1; off < 1024; off <<= 1) {
      int tv = (threadIdx.x >= off) ? buf[threadIdx.x - off] : 0;
      __syncthreads();
      buf[threadIdx.x] += tv;
      __syncthreads();
    }
    if (idx < N_NODES) prefix[idx] = carry + buf[threadIdx.x] - v;
    __syncthreads();
    if (threadIdx.x == 0) carry += buf[1023];
    __syncthreads();
  }
}

// ---------------- replicate jnp.repeat(nsum, counts): position-ordered denoms ----------------
__global__ void __launch_bounds__(256)
k_denom(const int* __restrict__ prefix, const int* __restrict__ cnt,
        const float* __restrict__ nsum, const float* __restrict__ esum,
        float* __restrict__ ndenom, float* __restrict__ edenom) {
  int j = blockIdx.x * 256 + threadIdx.x;
  if (j >= N_NODES) return;
  int start = prefix[j], c = cnt[j];
  float nv = nsum[j], ev = esum[j];
  for (int k = 0; k < c; ++k) { ndenom[start + k] = nv; edenom[start + k] = ev; }
}

// ---------------- normalize, scatter-add outputs, accumulate variance sums ----------------
__global__ void __launch_bounds__(256)
k_scatter(const int* __restrict__ eb, const float* __restrict__ hv,
          const float* __restrict__ evN, const float* __restrict__ natt,
          const float* __restrict__ eatt, const float* __restrict__ ndenom,
          const float* __restrict__ edenom, float* __restrict__ node_out,
          float* __restrict__ edge_out, double* __restrict__ accum) {
  __shared__ float red[32];
  int wave = threadIdx.x >> 5, lane = threadIdx.x & 31;
  int i = blockIdx.x * 8 + wave;   // 100000 blocks * 8 waves = 2E exactly
  int s, d;
  if (i < N_EDGES) { s = eb[2*i];            d = eb[2*i + 1]; }
  else { int ii = i - N_EDGES; s = eb[2*ii + 1]; d = eb[2*ii]; }
  float nn = natt[i] / ndenom[i];
  float en = eatt[i] / edenom[i];
  float4 h  = *(const float4*)(hv  + d * 128 + lane * 4);
  float4 ev = *(const float4*)(evN + d * 128 + lane * 4);
  float* no = node_out + s * 128 + lane * 4;
  float* eo = edge_out + s * 128 + lane * 4;
  atomicAdd(no + 0, h.x * nn); atomicAdd(no + 1, h.y * nn);
  atomicAdd(no + 2, h.z * nn); atomicAdd(no + 3, h.w * nn);
  atomicAdd(eo + 0, ev.x * en); atomicAdd(eo + 1, ev.y * en);
  atomicAdd(eo + 2, ev.z * en); atomicAdd(eo + 3, ev.w * en);
  if (lane == 0) {
    red[wave*4+0] = nn; red[wave*4+1] = nn*nn;
    red[wave*4+2] = en; red[wave*4+3] = en*en;
  }
  __syncthreads();
  if (threadIdx.x == 0) {
    double a0=0, a1=0, a2=0, a3=0;
    for (int w = 0; w < 8; ++w) {
      a0 += red[w*4+0]; a1 += red[w*4+1]; a2 += red[w*4+2]; a3 += red[w*4+3];
    }
    atomicAdd(accum+0, a0); atomicAdd(accum+1, a1);
    atomicAdd(accum+2, a2); atomicAdd(accum+3, a3);
  }
}

__global__ void k_finalize(const double* __restrict__ accum,
                           float* __restrict__ scalars) {
  double M = (double)TWO_E;
  scalars[0] = (float)((accum[1] - accum[0]*accum[0]/M) / (M - 1.0));
  scalars[1] = (float)((accum[3] - accum[2]*accum[2]/M) / (M - 1.0));
}

extern "C" void kernel_launch(void* const* d_in, const int* in_sizes, int n_in,
                              void* d_out, int out_size, void* d_ws, size_t ws_size,
                              hipStream_t stream) {
  const float* node_fts = (const float*)d_in[0];
  const float* edge_fts = (const float*)d_in[1];
  const int*   edges    = (const int*)d_in[2];
  const float* W_node   = (const float*)d_in[3];
  const float* W_edge   = (const float*)d_in[4];
  const float* a_node   = (const float*)d_in[5];
  const float* a_edge   = (const float*)d_in[6];

  char* ws = (char*)d_ws;
  float*  hv     = (float*) (ws + O_HV);
  float*  evN    = (float*) (ws + O_EV);
  float*  t      = (float*) (ws + O_T);
  float*  s1     = (float*) (ws + O_S1);
  float*  s2     = (float*) (ws + O_S2);
  float*  s3     = (float*) (ws + O_S3);
  float*  natt   = (float*) (ws + O_NATT);
  float*  eatt   = (float*) (ws + O_EATT);
  float*  ndenom = (float*) (ws + O_NDEN);
  float*  edenom = (float*) (ws + O_EDEN);
  float*  nsum   = (float*) (ws + O_NSUM);
  float*  esum   = (float*) (ws + O_ESUM);
  int*    cnt    = (int*)   (ws + O_CNT);
  double* accum  = (double*)(ws + O_ACC);
  int*    prefix = (int*)   (ws + O_PREF);

  float* out      = (float*)d_out;
  float* node_out = out;
  float* edge_out = out + (size_t)N_NODES * 128;
  float* scalars  = out + (size_t)2 * N_NODES * 128;

  hipMemsetAsync(d_out, 0, (size_t)out_size * sizeof(float), stream);
  hipMemsetAsync(ws + O_NSUM, 0, ZERO_BYTES, stream);

  k_gemm_node<<<391,   256, 0, stream>>>(node_fts, W_node, hv);
  k_gemm_edge<<<3125,  256, 0, stream>>>(edge_fts, W_edge, a_edge, evN, t);
  k_scores   <<<6250,  256, 0, stream>>>(hv, a_node, a_edge, s1, s2, s3);
  k_att      <<<3125,  256, 0, stream>>>(edges, s1, s2, s3, t, natt, eatt, nsum, esum, cnt);
  k_scan     <<<1,    1024, 0, stream>>>(cnt, prefix);
  k_denom    <<<196,   256, 0, stream>>>(prefix, cnt, nsum, esum, ndenom, edenom);
  k_scatter  <<<100000,256, 0, stream>>>(edges, hv, evN, natt, eatt, ndenom, edenom,
                                         node_out, edge_out, accum);
  k_finalize <<<1,       1, 0, stream>>>(accum, scalars);
}
